// Model_80041010528727
// MI455X (gfx1250) — matrix operand order repaired and checked by reading, not compiled, not run on hardware
//
#include <hip/hip_runtime.h>
#include <hip/hip_bf16.h>
#include <math.h>

// ---------------- types ----------------
typedef __attribute__((ext_vector_type(16))) __bf16          v16bf;
typedef __attribute__((ext_vector_type(8)))  float           v8f;
typedef __attribute__((ext_vector_type(8)))  unsigned short  v8u16;

union AFrag { v16bf v; v8u16 h[2]; };

#define B_   256
#define T_   512
#define HH   1024
#define NWG  32
#define NTHREADS 256
#define NWAVES (NWG * 8)          // 256 wave32s
#define LDS_STRIDE 1032           // 1024 + 8 pad: row-strided ds_load_b128 stays conflict-free
#define REGION (64 * LDS_STRIDE)  // one 64-row staging region (ushorts)
#define SMEM_BYTES (2 * REGION * 2)

__device__ __forceinline__ unsigned short f2bf(float f) {
  unsigned u = __float_as_uint(f);
  unsigned r = u + 0x7FFFu + ((u >> 16) & 1u);   // round-to-nearest-even
  return (unsigned short)(r >> 16);
}

// Branch-free tanh: 1 - 2/(1+exp(2v)).  v_exp_f32 + v_rcp_f32, ~1e-7 abs err,
// saturates correctly at +-1 (exp overflow/underflow).
__device__ __forceinline__ float tanh_fast(float v) {
  float e = __expf(2.0f * v);
  return 1.0f - 2.0f * __builtin_amdgcn_rcpf(1.0f + e);
}

// Device-scope grid barrier (generation based). Safe for repeated use.
__device__ __forceinline__ void grid_barrier(unsigned* cnt, unsigned* gen) {
  __syncthreads();
  if (threadIdx.x == 0) {
    __threadfence();
    unsigned g = __atomic_load_n(gen, __ATOMIC_RELAXED);
    unsigned arrived = __atomic_fetch_add(cnt, 1u, __ATOMIC_ACQ_REL);
    if (arrived == NWG - 1) {
      __atomic_store_n(cnt, 0u, __ATOMIC_RELAXED);   // reset BEFORE release
      __atomic_fetch_add(gen, 1u, __ATOMIC_RELEASE);
    } else {
      while (__atomic_load_n(gen, __ATOMIC_ACQUIRE) == g)
        __builtin_amdgcn_s_sleep(2);
    }
    __threadfence();
  }
  __syncthreads();
}

// fp32 -> bf16 weights (row-major bf16 rows feed the WMMA B-fragment layout
// directly: lane L<16 takes W[n0+L][k0..k0+15], lane L>=16 takes W[n0+L-16][k0+16..k0+31]).
__global__ void pack_bf16_kernel(const float* __restrict__ src,
                                 unsigned short* __restrict__ dst, int n) {
  int i = blockIdx.x * blockDim.x + threadIdx.x;
  if (i < n) dst[i] = f2bf(src[i]);
}

// Cooperative copy of 64 contiguous rows [64 x 1024] bf16 global -> LDS (padded rows).
__device__ __forceinline__ void stage_rows(unsigned short* __restrict__ dst,
                                           const unsigned short* __restrict__ src) {
#pragma unroll 4
  for (int c = threadIdx.x; c < 64 * 128; c += NTHREADS) {
    const int row = c >> 7, col8 = (c & 127) << 3;
    *(v8u16*)(dst + row * LDS_STRIDE + col8) = *(const v8u16*)(src + row * HH + col8);
  }
}

// 4 M-tiles x 1 N-tile, K=1024: A from LDS (64-row region), B from global (L2-resident
// bf16 weights).  One B-fragment feeds 4 WMMAs.
__device__ __forceinline__ void gemm4(const unsigned short* __restrict__ ldsA,
                                      const unsigned short* __restrict__ wbf,
                                      int n0, int lrow, int lhi, v8f acc[4]) {
  const unsigned short* brow = wbf + (size_t)(n0 + lrow) * HH + lhi * 16;
  const unsigned short* a0   = ldsA + lrow * LDS_STRIDE + lhi * 16;
#pragma unroll 2
  for (int k = 0; k < HH; k += 32) {
    v16bf bfrag = *(const v16bf*)(brow + k);
#pragma unroll
    for (int i = 0; i < 4; ++i) {
      AFrag a;
      const unsigned short* ar = a0 + i * (16 * LDS_STRIDE) + k;
      a.h[0] = *(const v8u16*)(ar);        // K = k..k+7   (k+16..k+23 for lanes>=16)
      a.h[1] = *(const v8u16*)(ar + 8);    // K = k+8..15  (k+24..k+31 for lanes>=16)
      acc[i] = __builtin_amdgcn_wmma_f32_16x16x32_bf16(
          false, a.v, false, bfrag, (short)0, acc[i], false, false);
    }
  }
}

__global__ __launch_bounds__(NTHREADS, 1)
void rnn_persistent(const float* __restrict__ x,        // [256,2,512]
                    const float* __restrict__ Wih1,     // [1024,2]
                    const float* __restrict__ bih1,
                    const float* __restrict__ bhh1,
                    const float* __restrict__ bih2,
                    const float* __restrict__ bhh2,
                    const float* __restrict__ Wlin,     // [2,1024]
                    const float* __restrict__ blin,
                    const unsigned short* __restrict__ W1bf,  // Whh1 bf16
                    const unsigned short* __restrict__ W2bf,  // Wih2 bf16
                    const unsigned short* __restrict__ W3bf,  // Whh2 bf16
                    unsigned short* __restrict__ h1bf,  // 2 x [256,1024] bf16 (zeroed)
                    unsigned short* __restrict__ h2bf,  // 2 x [256,1024] bf16 (zeroed)
                    float* __restrict__ h2f,            // 2 x [256,1024] f32
                    float* __restrict__ out,            // [256,2,512]
                    unsigned* __restrict__ sync) {
  extern __shared__ unsigned short lds[];
  unsigned short* R0 = lds;            // A region: h1 state (kept hot across the step)
  unsigned short* R1 = lds + REGION;   // A region: h2 state

  const int tid  = threadIdx.x;
  const int lane = tid & 31;
  const int wave = tid >> 5;
  const int gw   = blockIdx.x * 8 + wave;   // 0..255
  const int lrow = lane & 15;
  const int lhi  = lane >> 4;               // 0|1
  const int n0   = (gw & 63) << 4;          // N-strip of this wave
  const int mg   = blockIdx.x >> 3;         // 0..3, identical for all waves in WG
  const int rbase = mg << 6;                // 64-row block owned by this WG
  unsigned* cnt = sync;
  unsigned* gen = sync + 16;

  // Prologue: R0 <- h1 state (zeros). From then on R0 is refreshed during layer 2
  // with h1', which is exactly the next step's layer-1 A matrix.
  stage_rows(R0, h1bf + (size_t)rbase * HH);
  __syncthreads();

  for (int t = 0; t < T_; ++t) {
    const int cur = t & 1, nxt = cur ^ 1;
    unsigned short* h1w  = h1bf + (size_t)nxt * (B_ * HH);
    const unsigned short* h2r = h2bf + (size_t)cur * (B_ * HH);
    unsigned short* h2w  = h2bf + (size_t)nxt * (B_ * HH);
    float*          h2fw = h2f  + (size_t)nxt * (B_ * HH);

    // ---------- Layer 1: h1' = tanh(x_t Wih1^T + b + h1 Whh1^T) ----------
    {
      v8f acc[4] = {};
      gemm4(R0, W1bf, n0, lrow, lhi, acc);
      const int j = n0 + lrow;
      const float wi0 = Wih1[j * 2], wi1 = Wih1[j * 2 + 1];
      const float bs = bih1[j] + bhh1[j];
#pragma unroll
      for (int i = 0; i < 4; ++i) {
#pragma unroll
        for (int r = 0; r < 8; ++r) {
          const int b = rbase + i * 16 + (lhi << 3) + r;
          const float aff = x[(size_t)b * (2 * T_) + t] * wi0 +
                            x[(size_t)b * (2 * T_) + T_ + t] * wi1 + bs;
          h1w[(size_t)b * HH + j] = f2bf(tanh_fast(acc[i][r] + aff));
        }
      }
    }
    grid_barrier(cnt, gen);

    // Stage A matrices for layer 2 (and R0 doubles as next step's layer-1 A).
    stage_rows(R0, h1w + (size_t)rbase * HH);
    stage_rows(R1, h2r + (size_t)rbase * HH);
    __syncthreads();

    // ---------- Layer 2: h2' = tanh(h1' Wih2^T + h2 Whh2^T + b) ----------
    {
      v8f acc[4] = {};
      gemm4(R0, W2bf, n0, lrow, lhi, acc);
      gemm4(R1, W3bf, n0, lrow, lhi, acc);
      const int j = n0 + lrow;
      const float bs = bih2[j] + bhh2[j];
#pragma unroll
      for (int i = 0; i < 4; ++i) {
#pragma unroll
        for (int r = 0; r < 8; ++r) {
          const int b = rbase + i * 16 + (lhi << 3) + r;
          const float h = tanh_fast(acc[i][r] + bs);
          h2w[(size_t)b * HH + j] = f2bf(h);
          h2fw[(size_t)b * HH + j] = h;
        }
      }
    }
    grid_barrier(cnt, gen);

    // ---------- Output: out[b,c,t] = h2' . Wlin[c] + blin[c]  (wave gw -> b=gw) ----------
    {
      const int b = gw;
      const float* hv = h2fw + (size_t)b * HH;
      float s0 = 0.f, s1 = 0.f;
#pragma unroll 8
      for (int i = lane; i < HH; i += 32) {
        const float h = hv[i];
        s0 += h * Wlin[i];
        s1 += h * Wlin[HH + i];
      }
      for (int o = 16; o > 0; o >>= 1) {
        s0 += __shfl_down(s0, o, 32);
        s1 += __shfl_down(s1, o, 32);
      }
      if (lane == 0) {
        out[((size_t)b * 2 + 0) * T_ + t] = s0 + blin[0];
        out[((size_t)b * 2 + 1) * T_ + t] = s1 + blin[1];
      }
    }
    // out(t) reads h2f[nxt], first overwritten at step t+2 after two more grid
    // barriers -> no extra barrier needed here.
  }
}

extern "C" void kernel_launch(void* const* d_in, const int* in_sizes, int n_in,
                              void* d_out, int out_size, void* d_ws, size_t ws_size,
                              hipStream_t stream) {
  const float* x    = (const float*)d_in[0];
  const float* Wih1 = (const float*)d_in[1];
  const float* Whh1 = (const float*)d_in[2];
  const float* bih1 = (const float*)d_in[3];
  const float* bhh1 = (const float*)d_in[4];
  const float* Wih2 = (const float*)d_in[5];
  const float* Whh2 = (const float*)d_in[6];
  const float* bih2 = (const float*)d_in[7];
  const float* bhh2 = (const float*)d_in[8];
  const float* Wlin = (const float*)d_in[9];
  const float* blin = (const float*)d_in[10];
  // d_in[11] = future (reference uses 0)
  float* out = (float*)d_out;

  // ---- workspace layout (256B aligned blocks) ----
  size_t off = 0;
  auto take = [&](size_t bytes) { size_t o = off; off += (bytes + 255) & ~(size_t)255; return o; };
  const size_t sync_off = take(256);
  const size_t w1_off   = take((size_t)HH * HH * 2);
  const size_t w2_off   = take((size_t)HH * HH * 2);
  const size_t w3_off   = take((size_t)HH * HH * 2);
  const size_t h1_off   = take((size_t)2 * B_ * HH * 2);
  const size_t h2_off   = take((size_t)2 * B_ * HH * 2);
  const size_t h2f_off  = take((size_t)2 * B_ * HH * 4);
  const size_t total = off;

  char* ws = (char*)d_ws;
  hipMemsetAsync(d_ws, 0, total, stream);   // zero sync counters + h state

  unsigned short* W1bf = (unsigned short*)(ws + w1_off);
  unsigned short* W2bf = (unsigned short*)(ws + w2_off);
  unsigned short* W3bf = (unsigned short*)(ws + w3_off);
  unsigned short* h1bf = (unsigned short*)(ws + h1_off);
  unsigned short* h2bf = (unsigned short*)(ws + h2_off);
  float*          h2f  = (float*)(ws + h2f_off);
  unsigned*       sync = (unsigned*)(ws + sync_off);

  const int n = HH * HH;
  const int pb = (n + 255) / 256;
  pack_bf16_kernel<<<pb, 256, 0, stream>>>(Whh1, W1bf, n);
  pack_bf16_kernel<<<pb, 256, 0, stream>>>(Wih2, W2bf, n);
  pack_bf16_kernel<<<pb, 256, 0, stream>>>(Whh2, W3bf, n);

  rnn_persistent<<<NWG, NTHREADS, SMEM_BYTES, stream>>>(
      x, Wih1, bih1, bhh1, bih2, bhh2, Wlin, blin,
      W1bf, W2bf, W3bf, h1bf, h2bf, h2f, out, sync);

  (void)in_sizes; (void)n_in; (void)out_size; (void)ws_size;
}
